// Linear3Bit_57329223467740
// MI455X (gfx1250) — matrix operand order repaired and checked by reading, not compiled, not run on hardware
//
#include <hip/hip_runtime.h>

typedef __attribute__((ext_vector_type(16))) _Float16 v16h;
typedef __attribute__((ext_vector_type(8)))  _Float16 v8h;
typedef __attribute__((ext_vector_type(4)))  _Float16 v4h;
typedef __attribute__((ext_vector_type(8)))  float    v8f;
typedef __attribute__((ext_vector_type(4)))  float    v4f;

#define IN_F   4096
#define OUT_F  4096
#define MROWS  8192          // B*S = 4*2048

#define BM 128
#define BN 128
#define BK 64
#define LDA 72               // LDS row stride in halves (BK + 8, keeps b128 alignment)
#define LDB 72

#define WROW_BYTES (IN_F * 3 / 8)   // 1536 bytes per packed weight row

__global__ __launch_bounds__(256)
void linear3bit_wmma_f16(const float* __restrict__ x,
                         const unsigned char* __restrict__ wp,
                         const _Float16* __restrict__ scale,
                         const _Float16* __restrict__ bias,
                         float* __restrict__ y)
{
    __shared__ _Float16 As[BM * LDA];   // 18 KB
    __shared__ _Float16 Bs[BN * LDB];   // 18 KB

    const int tid  = threadIdx.x;
    const int lane = tid & 31;
    const int l16  = lane & 15;
    const int lhi  = lane >> 4;         // 0 = low half-wave, 1 = high half-wave
    const int wave = tid >> 5;
    const int wm0  = (wave >> 2) * 64;  // wave M origin inside block: 0 / 64
    const int wn0  = (wave & 3) * 32;   // wave N origin inside block: 0..96

    const int n0 = blockIdx.x * BN;
    const int m0 = blockIdx.y * BM;

    // ---- weight-dequant thread mapping: 2 threads per output row chunk ----
    const int nloc  = tid >> 1;                         // 0..127 (local n row)
    const int khalf = tid & 1;                          // which 32-wide half of BK
    const int gn_w  = n0 + nloc;
    const float srow = (float)scale[gn_w] * (1.0f / 3.0f);
    const unsigned char* wrow = wp + (size_t)gn_w * WROW_BYTES + khalf * 12;

    v8f acc[4][2];
    #pragma unroll
    for (int i = 0; i < 4; ++i)
        #pragma unroll
        for (int j = 0; j < 2; ++j)
            acc[i][j] = {};

    const int KT = IN_F / BK;           // 64 K-iterations
    for (int kt = 0; kt < KT; ++kt) {
        __syncthreads();                // previous iter's LDS reads done

        // ---- stage A tile: BM x BK f32 -> f16 in LDS ----
        const float* xbase = x + (size_t)m0 * IN_F + (size_t)kt * BK;
        #pragma unroll
        for (int it = 0; it < 8; ++it) {
            int e   = tid + it * 256;   // float4 chunk id, 0..2047
            int row = e >> 4;           // 16 float4 per row
            int c4  = e & 15;
            v4f v = *(const v4f*)(xbase + (size_t)row * IN_F + c4 * 4);
            v4h h;
            h[0] = (_Float16)v.x; h[1] = (_Float16)v.y;
            h[2] = (_Float16)v.z; h[3] = (_Float16)v.w;
            *(v4h*)&As[row * LDA + c4 * 4] = h;
        }

        // ---- stage B tile: dequant 3-bit codes -> f16, LDS layout [n][k] ----
        {
            const unsigned char* p = wrow + (size_t)kt * 24;   // 24 B = 64 codes/row
            unsigned w0 = *(const unsigned*)(p);
            unsigned w1 = *(const unsigned*)(p + 4);
            unsigned w2 = *(const unsigned*)(p + 8);
            unsigned b24[4];
            b24[0] =  w0 & 0xFFFFFFu;
            b24[1] = (w0 >> 24) | (w1 << 8);
            b24[2] = (w1 >> 16) | (w2 << 16);
            b24[3] =  w2 >> 8;
            #pragma unroll
            for (int g = 0; g < 4; ++g) {
                unsigned b = b24[g];
                v8h h;
                #pragma unroll
                for (int jj = 0; jj < 8; ++jj) {
                    int c = (int)((b >> (3 * jj)) & 7u);       // code in [0,6]
                    h[jj] = (_Float16)((float)(c - 3) * srow);
                }
                *(v8h*)&Bs[nloc * LDB + khalf * 32 + g * 8] = h;
            }
        }

        __syncthreads();

        // prefetch next A tile (whole line per address; exercises global_prefetch_b8)
        if (kt + 1 < KT) {
            const float* nx = x + (size_t)m0 * IN_F + (size_t)(kt + 1) * BK;
            #pragma unroll
            for (int it = 0; it < 8; ++it) {
                int e = tid + it * 256;
                __builtin_prefetch(nx + (size_t)(e >> 4) * IN_F + (e & 15) * 4, 0, 1);
            }
        }

        // ---- compute: 2 K-substeps of 32, 4x2 WMMA tiles per wave ----
        #pragma unroll
        for (int ks = 0; ks < BK; ks += 32) {
            // B fragments: lane holds col n=l16 (+0/+16 K for half-waves),
            // 16 contiguous halves -> two ds_load_b128
            union { v16h v; v8h h[2]; } bf[2];
            #pragma unroll
            for (int j = 0; j < 2; ++j) {
                const _Float16* bp = &Bs[(wn0 + 16 * j + l16) * LDB + ks + lhi * 16];
                bf[j].h[0] = *(const v8h*)(bp);
                bf[j].h[1] = *(const v8h*)(bp + 8);
            }
            #pragma unroll
            for (int i = 0; i < 4; ++i) {
                // A fragment: lane holds row m, K runs {kb..kb+15},
                // kb = ks + 16*lhi -> two ds_load_b128
                union { v16h v; v8h h[2]; } af;
                const _Float16* ap = &As[(wm0 + 16 * i + l16) * LDA + ks + lhi * 16];
                af.h[0] = *(const v8h*)(ap);
                af.h[1] = *(const v8h*)(ap + 8);
                #pragma unroll
                for (int j = 0; j < 2; ++j) {
                    acc[i][j] = __builtin_amdgcn_wmma_f32_16x16x32_f16(
                        false, af.v, false, bf[j].v,
                        (short)0, acc[i][j], false, false);
                }
            }
        }
    }

    // ---- epilogue: bias add + store (C/D layout: VGPR r -> row r + 8*lhi) ----
    #pragma unroll
    for (int j = 0; j < 2; ++j) {
        int gcol = n0 + wn0 + 16 * j + l16;
        float bv = (float)bias[gcol];
        #pragma unroll
        for (int i = 0; i < 4; ++i) {
            int grow = m0 + wm0 + 16 * i + lhi * 8;
            float* out = y + (size_t)grow * OUT_F + gcol;
            #pragma unroll
            for (int r = 0; r < 8; ++r)
                out[(size_t)r * OUT_F] = acc[i][j][r] + bv;
        }
    }
}

extern "C" void kernel_launch(void* const* d_in, const int* in_sizes, int n_in,
                              void* d_out, int out_size, void* d_ws, size_t ws_size,
                              hipStream_t stream) {
    (void)in_sizes; (void)n_in; (void)out_size; (void)d_ws; (void)ws_size;
    const float*          x     = (const float*)d_in[0];
    const unsigned char*  wpack = (const unsigned char*)d_in[1];
    const _Float16*       scale = (const _Float16*)d_in[2];
    const _Float16*       bias  = (const _Float16*)d_in[3];
    float*                y     = (float*)d_out;

    dim3 grid(OUT_F / BN, MROWS / BM);   // 32 x 64 = 2048 workgroups
    dim3 block(256);                      // 8 waves (wave32)
    linear3bit_wmma_f16<<<grid, block, 0, stream>>>(x, wpack, scale, bias, y);
}
